// BlockGNN_64080912056838
// MI455X (gfx1250) — matrix operand order repaired and checked by reading, not compiled, not run on hardware
//
#include <hip/hip_runtime.h>
#include <hip/hip_bf16.h>

// GCN: 4x { h = act @ W (WMMA f16), out = scatter(norm * h[src] -> dst) + selfloop + bias, ReLU }
// then mean-pool over 512 graphs and a tiny 128x10 head.
// MI455X reasoning: whole per-layer working set (~130MB) fits in 192MB L2, so the
// edge scatter uses global_atomic_add_f32 (L2-resident); dense X@W uses
// v_wmma_f32_16x16x32_f16 tiles.

#define HID 128
#define NCLS 10
#define NGRAPHS 512

typedef __attribute__((ext_vector_type(16))) _Float16 v16h;
typedef __attribute__((ext_vector_type(8)))  float    v8f;

static __host__ unsigned nblk(long long n, int b) { return (unsigned)((n + b - 1) / b); }

// ---------------- prep ----------------

__global__ void k_zero(float* p, long long n) {
  long long i = (long long)blockIdx.x * blockDim.x + threadIdx.x;
  if (i < n) p[i] = 0.0f;
}

__global__ void k_count_deg(const int* __restrict__ dst, float* __restrict__ deg, int E) {
  int e = blockIdx.x * blockDim.x + threadIdx.x;
  if (e < E) atomicAdd(&deg[dst[e]], 1.0f);
}

// deg -> dinv in place; +1 accounts for the self-loop edge
__global__ void k_dinv(float* __restrict__ d, int N) {
  int i = blockIdx.x * blockDim.x + threadIdx.x;
  if (i < N) {
    float dg = d[i] + 1.0f;
    d[i] = rsqrtf(dg);
  }
}

__global__ void k_edge_norm(const int* __restrict__ src, const int* __restrict__ dst,
                            const float* __restrict__ dinv, float* __restrict__ norm, int E) {
  int e = blockIdx.x * blockDim.x + threadIdx.x;
  if (e < E) norm[e] = dinv[src[e]] * dinv[dst[e]];
}

// W[k][n] (f32) -> Wt[n][k] (f16), 128x128
__global__ void k_convert_wt(const float* __restrict__ W, _Float16* __restrict__ Wt) {
  int t = blockIdx.x * blockDim.x + threadIdx.x;
  if (t < HID * HID) {
    int n = t >> 7, k = t & 127;
    Wt[n * HID + k] = (_Float16)W[k * HID + n];
  }
}

// ---------------- dense: H = act @ W via WMMA ----------------
// grid.x = M/16 (m-tile), block = 256 = 8 waves, wave w owns n-tile w (16 cols).
// K = 128, unrolled into 4 x v_wmma_f32_16x16x32_f16.

struct F8v { float4 a, b; };

__global__ __launch_bounds__(256)
void k_gemm_wmma(const float* __restrict__ A, const _Float16* __restrict__ Wt,
                 float* __restrict__ C) {
  const int lane = threadIdx.x & 31;
  const int wave = threadIdx.x >> 5;
  const int h    = lane >> 4;          // half-wave
  const int ln16 = lane & 15;
  const int mt   = blockIdx.x;
  const int row  = mt * 16 + ln16;     // A row for this lane
  const int n0   = wave * 16;
  const int ncol = n0 + ln16;          // B column for this lane

  const float*    ap = A  + (size_t)row  * HID;
  const _Float16* bp = Wt + (size_t)ncol * HID + 8 * h;

  v8f acc = {};
#pragma unroll
  for (int kk = 0; kk < 4; ++kk) {
    const int kb = kk * 32;
    // A: two contiguous 8-float chunks at K = kb+8h and K = kb+16+8h
    F8v lo = *(const F8v*)(ap + kb + 8 * h);
    F8v hi = *(const F8v*)(ap + kb + 16 + 8 * h);
    const float* lof = (const float*)&lo;
    const float* hif = (const float*)&hi;
    v16h a;
#pragma unroll
    for (int j = 0; j < 8; ++j) {
      a[j]     = (_Float16)lof[j];
      a[8 + j] = (_Float16)hif[j];
    }
    // B: two 8-f16 chunks at K = kb+8h and K = kb+16+8h from transposed weights
    typedef __attribute__((ext_vector_type(8))) _Float16 v8h_;
    v8h_ blo = *(const v8h_*)(bp + kb);
    v8h_ bhi = *(const v8h_*)(bp + kb + 16);
    v16h b;
#pragma unroll
    for (int j = 0; j < 8; ++j) {
      b[j]     = blo[j];
      b[8 + j] = bhi[j];
    }
    acc = __builtin_amdgcn_wmma_f32_16x16x32_f16(false, a, false, b,
                                                 (short)0, acc, false, false);
  }
  // D layout: VGPR r -> M = r + 8h, N = lane&15
  float* cp = C + (size_t)(mt * 16 + 8 * h) * HID + n0 + ln16;
#pragma unroll
  for (int r = 0; r < 8; ++r) cp[(size_t)r * HID] = acc[r];
}

// ---------------- sparse aggregate ----------------

// out[i][f] = h[i][f] * dinv[i]^2 (self loop) + bias[f]
__global__ void k_init_self(const float* __restrict__ h, const float* __restrict__ dinv,
                            const float* __restrict__ bias, float* __restrict__ out,
                            long long NF) {
  long long t = (long long)blockIdx.x * blockDim.x + threadIdx.x;
  if (t < NF) {
    int i = (int)(t >> 7), f = (int)(t & 127);
    float di = dinv[i];
    out[t] = h[t] * di * di + bias[f];
  }
}

// edge-parallel scatter-add; resolves in 192MB L2 (h/out both resident)
__global__ void k_scatter(const float* __restrict__ h, const int* __restrict__ src,
                          const int* __restrict__ dst, const float* __restrict__ norm,
                          float* __restrict__ out, long long EF) {
  long long t = (long long)blockIdx.x * blockDim.x + threadIdx.x;
  if (t < EF) {
    int e = (int)(t >> 7), f = (int)(t & 127);
    float v = h[(size_t)src[e] * HID + f] * norm[e];
    atomicAdd(&out[(size_t)dst[e] * HID + f], v);
  }
}

__global__ void k_relu(float* __restrict__ p, long long n) {
  long long i = (long long)blockIdx.x * blockDim.x + threadIdx.x;
  if (i < n) p[i] = fmaxf(p[i], 0.0f);
}

// ---------------- pooling + head ----------------

__global__ void k_count_batch(const int* __restrict__ batch, float* __restrict__ counts, int N) {
  int i = blockIdx.x * blockDim.x + threadIdx.x;
  if (i < N) atomicAdd(&counts[batch[i]], 1.0f);
}

__global__ void k_pool(const float* __restrict__ act, const int* __restrict__ batch,
                       float* __restrict__ pooled, long long NF) {
  long long t = (long long)blockIdx.x * blockDim.x + threadIdx.x;
  if (t < NF) {
    int i = (int)(t >> 7), f = (int)(t & 127);
    atomicAdd(&pooled[(size_t)batch[i] * HID + f], act[t]);
  }
}

__global__ void k_mean(const float* __restrict__ pooled, const float* __restrict__ counts,
                       float* __restrict__ gm) {
  int t = blockIdx.x * blockDim.x + threadIdx.x;
  if (t < NGRAPHS * HID) {
    int g = t >> 7;
    gm[t] = pooled[t] / fmaxf(counts[g], 1.0f);
  }
}

__global__ void k_head(const float* __restrict__ gm, const float* __restrict__ Wl,
                       const float* __restrict__ bl, float* __restrict__ y) {
  int t = blockIdx.x * blockDim.x + threadIdx.x;
  if (t < NGRAPHS * NCLS) {
    int g = t / NCLS, c = t % NCLS;
    float acc = bl[c];
    const float* gp = gm + (size_t)g * HID;
#pragma unroll 8
    for (int k = 0; k < HID; ++k) acc += gp[k] * Wl[k * NCLS + c];
    y[t] = acc;
  }
}

// ---------------- driver ----------------

extern "C" void kernel_launch(void* const* d_in, const int* in_sizes, int n_in,
                              void* d_out, int out_size, void* d_ws, size_t ws_size,
                              hipStream_t stream) {
  const float* x     = (const float*)d_in[0];
  const int*   ei    = (const int*)d_in[1];
  const int*   batch = (const int*)d_in[2];
  const float* Ws[4] = {(const float*)d_in[3], (const float*)d_in[5],
                        (const float*)d_in[7], (const float*)d_in[9]};
  const float* Bs[4] = {(const float*)d_in[4], (const float*)d_in[6],
                        (const float*)d_in[8], (const float*)d_in[10]};
  const float* Wl = (const float*)d_in[11];
  const float* bl = (const float*)d_in[12];

  const int N = in_sizes[0] / HID;       // 100000 (multiple of 16)
  const int E = in_sizes[1] / 2;         // 1600000
  const int* src = ei;
  const int* dst = ei + E;

  // workspace carve-out (~110 MB)
  char*  wsp = (char*)d_ws;
  size_t off = 0;
  auto take = [&](size_t bytes) -> void* {
    void* p = wsp + off;
    off = (off + bytes + 255) & ~(size_t)255;
    return p;
  };
  float*    dinv   = (float*)take((size_t)N * 4);            // deg then dinv (in place)
  float*    norm   = (float*)take((size_t)E * 4);
  float*    act    = (float*)take((size_t)N * HID * 4);      // activations (scatter target)
  float*    H      = (float*)take((size_t)N * HID * 4);      // GEMM output
  _Float16* Wt     = (_Float16*)take((size_t)HID * HID * 2); // transposed f16 weights
  float*    pooled = (float*)take((size_t)NGRAPHS * HID * 4);
  float*    counts = (float*)take((size_t)NGRAPHS * 4);

  const long long NF = (long long)N * HID;
  const long long EF = (long long)E * HID;

  // degrees -> dinv -> per-edge norms
  k_zero<<<nblk(N, 256), 256, 0, stream>>>(dinv, N);
  k_count_deg<<<nblk(E, 256), 256, 0, stream>>>(dst, dinv, E);
  k_dinv<<<nblk(N, 256), 256, 0, stream>>>(dinv, N);
  k_edge_norm<<<nblk(E, 256), 256, 0, stream>>>(src, dst, dinv, norm, E);

  // 4 GCN layers
  for (int L = 0; L < 4; ++L) {
    const float* in = (L == 0) ? x : act;
    k_convert_wt<<<nblk(HID * HID, 256), 256, 0, stream>>>(Ws[L], Wt);
    k_gemm_wmma<<<N / 16, 256, 0, stream>>>(in, Wt, H);
    k_init_self<<<nblk(NF, 256), 256, 0, stream>>>(H, dinv, Bs[L], act, NF);
    k_scatter<<<nblk(EF, 256), 256, 0, stream>>>(H, src, dst, norm, act, EF);
    k_relu<<<nblk(NF, 256), 256, 0, stream>>>(act, NF);
  }

  // mean pool + head
  float* y  = (float*)d_out;                 // [512 x 10]
  float* gm = y + NGRAPHS * NCLS;            // [512 x 128]
  k_zero<<<nblk(NGRAPHS * HID, 256), 256, 0, stream>>>(pooled, NGRAPHS * HID);
  k_zero<<<nblk(NGRAPHS, 256), 256, 0, stream>>>(counts, NGRAPHS);
  k_count_batch<<<nblk(N, 256), 256, 0, stream>>>(batch, counts, N);
  k_pool<<<nblk(NF, 256), 256, 0, stream>>>(act, batch, pooled, NF);
  k_mean<<<nblk(NGRAPHS * HID, 256), 256, 0, stream>>>(pooled, counts, gm);
  k_head<<<nblk(NGRAPHS * NCLS, 256), 256, 0, stream>>>(gm, Wl, bl, y);
}
